// ContrastiveLossELI5_6365141533000
// MI455X (gfx1250) — matrix operand order repaired and checked by reading, not compiled, not run on hardware
//
#include <hip/hip_runtime.h>
#include <hip/hip_bf16.h>
#include <math.h>

typedef __attribute__((ext_vector_type(16))) _Float16 v16h;
typedef __attribute__((ext_vector_type(8)))  _Float16 v8h;
typedef __attribute__((ext_vector_type(4)))  _Float16 v4h;
typedef __attribute__((ext_vector_type(8)))  float    v8f;

#define NROWS 4096
#define DIMK  128
#define TOT   8192                       // 2N
#define CHUNKS 4
#define CHUNK_COLS (TOT / CHUNKS)        // 2048 columns per wave
#define TILES_PER_CHUNK (CHUNK_COLS / 16)// 128 column tiles per chunk
#define ROWS_PER_WAVE 32                 // two 16-row A sets share each B fragment
#define INV_TEMP 2.0f
// exp(INV_TEMP * c) == exp2(c * INV_TEMP * log2(e)); args are in [-6,6] so the
// raw v_exp_f32 (no denormal fixup) is exact enough -> __builtin_amdgcn_exp2f.
#define EXP2_SCALE 2.88539008177792681472f
#define FAST_EXP2(x) __builtin_amdgcn_exp2f(x)

// ---------------- Kernel 1: L2-normalize rows, emit f16 z [8192][128] ----------------
__global__ void nrm_kernel(const float* __restrict__ ei, const float* __restrict__ ej,
                           _Float16* __restrict__ zh) {
  const int lane = threadIdx.x & 31;
  const int wv   = threadIdx.x >> 5;
  const int row  = blockIdx.x * 8 + wv;                       // 8 waves per block, 1 row per wave
  const float* src = (row < NROWS) ? (ei + (size_t)row * DIMK)
                                   : (ej + (size_t)(row - NROWS) * DIMK);
  float4 x = ((const float4*)src)[lane];                      // 32 lanes * 4 = 128
  float ss = x.x*x.x + x.y*x.y + x.z*x.z + x.w*x.w;
  #pragma unroll
  for (int msk = 16; msk >= 1; msk >>= 1) ss += __shfl_xor(ss, msk, 32);
  const float r = rsqrtf(ss);
  v4h o;
  o[0] = (_Float16)(x.x * r); o[1] = (_Float16)(x.y * r);
  o[2] = (_Float16)(x.z * r); o[3] = (_Float16)(x.w * r);
  ((v4h*)(zh + (size_t)row * DIMK))[lane] = o;
}

__device__ __forceinline__ void load_a_frags(const _Float16* __restrict__ arow,
                                             int hf, v16h a[4]) {
  // A fragments (16x32 f16 layout): halves 0..7 -> K = 32s + hf*16 + i ;
  // halves 8..15 -> K = 32s + hf*16 + 8 + i
  #pragma unroll
  for (int s = 0; s < 4; ++s) {
    const _Float16* p = arow + 32 * s + hf * 16;
    v8h lo = *(const v8h*)p;
    v8h hi = *(const v8h*)(p + 8);
    v16h t;
    #pragma unroll
    for (int i = 0; i < 8; ++i) { t[i] = lo[i]; t[i + 8] = hi[i]; }
    a[s] = t;
  }
}

// ------- Kernel 2: fused 16x16 WMMA tiles + streaming sum-of-exp per row -------
// logits = 2*cos(theta) in [-2,2]; exp() cannot overflow, so no running max:
// lse = log(sum exp(x)) directly, all merges are plain sums.
// Diagonal and positive-pair columns are tile-aligned per wave, so their
// handling lives in a rarely-taken uniform-branch fix-up, keeping the hot
// epilogue to mul + exp + accumulate.
// Block = 128 threads = 4 waves; blockIdx.x = 32-row block; wave id = column chunk.
__global__ void __launch_bounds__(128, 1)
lse_kernel(const _Float16* __restrict__ zh,
           float* __restrict__ ps, float* __restrict__ pp) {
  const int lane  = threadIdx.x & 31;
  const int chunk = threadIdx.x >> 5;          // 0..3
  const int rbase = blockIdx.x * ROWS_PER_WAVE;
  const int mrow  = lane & 15;
  const int hf    = lane >> 4;

  v16h a0[4], a1[4];
  load_a_frags(zh + (size_t)(rbase      + mrow) * DIMK, hf, a0);
  load_a_frags(zh + (size_t)(rbase + 16 + mrow) * DIMK, hf, a1);

  float sacc0[8], pos0[8], sacc1[8], pos1[8];
  #pragma unroll
  for (int r = 0; r < 8; ++r) {
    sacc0[r] = 0.0f; pos0[r] = 0.0f;
    sacc1[r] = 0.0f; pos1[r] = 0.0f;
  }

  // C layout: lanes 0-15 -> M=0..7, lanes 16-31 -> M=8..15
  const int g0 = rbase + hf * 8;        // rows for c0
  const int g1 = g0 + 16;               // rows for c1

  // B fragment (32x16 f16 layout): lane holds column cbase+mrow, K = 32s + hf*16 + (0..15).
  // Ping-pong double buffering (no register rotation copies).
  const _Float16* bcol = zh + (size_t)(chunk * CHUNK_COLS + mrow) * DIMK + hf * 16;
  const size_t tile_stride = (size_t)16 * DIMK;

  v16h bufA[4], bufB[4];
  #pragma unroll
  for (int s = 0; s < 4; ++s) bufA[s] = *(const v16h*)(bcol + 32 * s);
  #pragma unroll
  for (int s = 0; s < 4; ++s) bufB[s] = *(const v16h*)(bcol + tile_stride + 32 * s);

  for (int ct = 0; ct < TILES_PER_CHUNK; ct += 2) {
    #pragma unroll
    for (int half_it = 0; half_it < 2; ++half_it) {
      const v16h* bcur = half_it ? bufB : bufA;
      v16h*       bpre = half_it ? bufA : bufB;   // buffer just freed -> prefetch into it

      // Unconditional prefetch of tile (ct + half_it + 2), clamped at the end so
      // the last iterations just re-load a resident tile (no branchy load block).
      {
        int pidx = ct + half_it + 2;
        pidx = (pidx < TILES_PER_CHUNK) ? pidx : (TILES_PER_CHUNK - 1);
        const _Float16* pnext = bcol + (size_t)pidx * tile_stride;
        #pragma unroll
        for (int s = 0; s < 4; ++s) bpre[s] = *(const v16h*)(pnext + 32 * s);
      }

      // Two independent accumulator chains sharing each B fragment.
      v8f c0 = {}, c1 = {};
      #pragma unroll
      for (int s = 0; s < 4; ++s) {
        c0 = __builtin_amdgcn_wmma_f32_16x16x32_f16(false, a0[s], false, bcur[s], (short)0, c0, false, false);
        c1 = __builtin_amdgcn_wmma_f32_16x16x32_f16(false, a1[s], false, bcur[s], (short)0, c1, false, false);
      }

      // hot epilogue: one v_mul + one v_exp_f32 + one add per element
      #pragma unroll
      for (int r = 0; r < 8; ++r) {
        sacc0[r] += FAST_EXP2(c0[r] * EXP2_SCALE);
        sacc1[r] += FAST_EXP2(c1[r] * EXP2_SCALE);
      }

      // Uniform-branch fix-up: diagonal columns live only in tiles
      // cbase == rbase / rbase+16; positive-pair columns only in
      // cbase == rbase^4096 / (rbase+16)^4096 (4096 is tile-aligned).
      const int cbase = chunk * CHUNK_COLS + (ct + half_it) * 16;
      const bool f_d0 = (cbase == rbase);
      const bool f_d1 = (cbase == rbase + 16);
      const bool f_p0 = (cbase == (rbase ^ NROWS));
      const bool f_p1 = (cbase == ((rbase + 16) ^ NROWS));
      if (f_d0 | f_d1 | f_p0 | f_p1) {
        const int gcol = cbase + mrow;
        #pragma unroll
        for (int r = 0; r < 8; ++r) {
          const int grow0 = g0 + r;
          const int grow1 = g1 + r;
          // subtract the *identical* exp2 expression -> exact cancellation
          if (f_d0) sacc0[r] -= (gcol == grow0) ? FAST_EXP2(c0[r] * EXP2_SCALE) : 0.0f;
          if (f_d1) sacc1[r] -= (gcol == grow1) ? FAST_EXP2(c1[r] * EXP2_SCALE) : 0.0f;
          if (f_p0) pos0[r]  += (gcol == (grow0 ^ NROWS)) ? (INV_TEMP * c0[r]) : 0.0f;
          if (f_p1) pos1[r]  += (gcol == (grow1 ^ NROWS)) ? (INV_TEMP * c1[r]) : 0.0f;
        }
      }
    }
  }

  // additive merge across the 16 lanes holding the same rows
  #pragma unroll
  for (int r = 0; r < 8; ++r) {
    #pragma unroll
    for (int msk = 1; msk < 16; msk <<= 1) {
      sacc0[r] += __shfl_xor(sacc0[r], msk, 32);
      pos0[r]  += __shfl_xor(pos0[r],  msk, 32);
      sacc1[r] += __shfl_xor(sacc1[r], msk, 32);
      pos1[r]  += __shfl_xor(pos1[r],  msk, 32);
    }
  }
  if (mrow == 0) {
    #pragma unroll
    for (int r = 0; r < 8; ++r) {
      const int i0 = chunk * TOT + g0 + r;
      ps[i0] = sacc0[r];  pp[i0] = pos0[r];
      const int i1 = chunk * TOT + g1 + r;
      ps[i1] = sacc1[r];  pp[i1] = pos1[r];
    }
  }
}

// ---------------- Kernel 3: merge chunk partials, mean(log(S) - pos) ----------------
__global__ void fin_kernel(const float* __restrict__ ps, const float* __restrict__ pp,
                           float* __restrict__ out) {
  __shared__ float red[256];
  float acc = 0.0f;
  for (int row = threadIdx.x; row < TOT; row += 256) {
    float S = 0.0f, P = 0.0f;
    #pragma unroll
    for (int c = 0; c < CHUNKS; ++c) {
      S += ps[c * TOT + row];
      P += pp[c * TOT + row];
    }
    acc += logf(S) - P;
  }
  red[threadIdx.x] = acc;
  __syncthreads();
  for (int s = 128; s > 0; s >>= 1) {
    if ((int)threadIdx.x < s) red[threadIdx.x] += red[threadIdx.x + s];
    __syncthreads();
  }
  if (threadIdx.x == 0) out[0] = red[0] / (float)TOT;
}

extern "C" void kernel_launch(void* const* d_in, const int* in_sizes, int n_in,
                              void* d_out, int out_size, void* d_ws, size_t ws_size,
                              hipStream_t stream) {
  (void)in_sizes; (void)n_in; (void)out_size; (void)ws_size;
  const float* ei = (const float*)d_in[0];
  const float* ej = (const float*)d_in[1];

  _Float16* zh = (_Float16*)d_ws;                                  // 8192*128*2 = 2 MB
  float* ps = (float*)((char*)d_ws + (size_t)TOT * DIMK * sizeof(_Float16));
  float* pp = ps + (size_t)CHUNKS * TOT;                           // 2 * 128 KB partials

  nrm_kernel<<<TOT / 8, 256, 0, stream>>>(ei, ej, zh);
  lse_kernel<<<TOT / ROWS_PER_WAVE, 32 * CHUNKS, 0, stream>>>(zh, ps, pp);
  fin_kernel<<<1, 256, 0, stream>>>(ps, pp, (float*)d_out);
}
